// PeepholeLSTM_5171140625007
// MI455X (gfx1250) — matrix operand order repaired and checked by reading, not compiled, not run on hardware
//
#include <hip/hip_runtime.h>
#include <cstdint>
#include <cstddef>

// ---------------- types ----------------
typedef __bf16 bf16_t;
typedef __attribute__((ext_vector_type(8)))  __bf16 v8bf;
typedef __attribute__((ext_vector_type(16))) __bf16 v16bf;
typedef __attribute__((ext_vector_type(8)))  float  v8f;

// ---------------- problem constants ----------------
constexpr int Bsz = 256;   // batch
constexpr int Tsz = 128;   // time steps
constexpr int Fsz = 512;   // input features
constexpr int Hsz = 1024;  // hidden
constexpr int G4  = 4096;  // 4*H
constexpr int KC  = 1536;  // F + H (combined K for g-GEMM)

// ---------------- fragment loaders ----------------
// A fragment (16-bit 16x32, k order of this code, not the ISA table): lane m = lane&15,
// kh = (lane>>4)*16 ; elements: K[kh..kh+15] contiguous
__device__ __forceinline__ v16bf ld_fragA(const bf16_t* p) {
    union { v16bf w; v8bf h[2]; } u;
    u.h[0] = *(const v8bf*)(p);
    u.h[1] = *(const v8bf*)(p + 8);
    return u.w;
}
// B fragment (16-bit 32x16): lane n = lane&15, kb = (lane>>4)*16 ;
// 16 contiguous K values from B^T stored row-major [N,K]
__device__ __forceinline__ v16bf ld_fragB(const bf16_t* p) {
    union { v16bf w; v8bf h[2]; } u;
    u.h[0] = *(const v8bf*)(p);
    u.h[1] = *(const v8bf*)(p + 8);
    return u.w;
}

__device__ __forceinline__ float sigmoidf_(float x) {
    return 1.0f / (1.0f + __expf(-x));
}

#define WMMA_BF16(a, b, c) \
    __builtin_amdgcn_wmma_f32_16x16x32_bf16(false, (a), false, (b), (short)0, (c), false, false)

// =====================================================================
// Kernel G: g[B,4H] = [x_t | h] @ Wcomb + bias    (K = 1536, bf16 WMMA)
// block = 128 thr (4 waves in 2x2); wave tile 32x32; block tile 64x64
// grid = (B/64, 4H/64) = (4, 64).  Double-buffered k-pipeline.
// =====================================================================
__global__ __launch_bounds__(128) void lstm_kg(
    const bf16_t* __restrict__ xb,   // [B, T, F] bf16
    const bf16_t* __restrict__ hB,   // [B, H]    bf16 (h_{t-1})
    const bf16_t* __restrict__ Wt,   // [4H, KC]  bf16 (Wcomb^T)
    const float*  __restrict__ bias, // [4H]
    float*        __restrict__ g,    // [B, 4H]
    int t)
{
    const int lane = threadIdx.x & 31;
    const int wave = threadIdx.x >> 5;
    const int m0   = blockIdx.x * 64 + (wave & 1) * 32;
    const int n0   = blockIdx.y * 64 + (wave >> 1) * 32;
    const int nl   = lane & 15;
    const int khA  = (lane >> 4) * 16;
    const int kbB  = (lane >> 4) * 16;

    const bf16_t* xr0 = xb + ((size_t)(m0 + nl) * Tsz + t) * Fsz + khA;
    const bf16_t* xr1 = xb + ((size_t)(m0 + 16 + nl) * Tsz + t) * Fsz + khA;
    const bf16_t* hr0 = hB + (size_t)(m0 + nl) * Hsz + khA;
    const bf16_t* hr1 = hB + (size_t)(m0 + 16 + nl) * Hsz + khA;
    const bf16_t* bc0 = Wt + (size_t)(n0 + nl) * KC + kbB;
    const bf16_t* bc1 = Wt + (size_t)(n0 + 16 + nl) * KC + kbB;

    v8f acc[2][2] = {};
    v16bf a[2][2], b[2][2];   // [buf][sub]

    auto loadF = [&](int buf, int kc) {
        if (kc < Fsz) {
            a[buf][0] = ld_fragA(xr0 + kc);
            a[buf][1] = ld_fragA(xr1 + kc);
        } else {
            a[buf][0] = ld_fragA(hr0 + (kc - Fsz));
            a[buf][1] = ld_fragA(hr1 + (kc - Fsz));
        }
        b[buf][0] = ld_fragB(bc0 + kc);
        b[buf][1] = ld_fragB(bc1 + kc);
    };
    auto mmac = [&](int buf) {
        acc[0][0] = WMMA_BF16(a[buf][0], b[buf][0], acc[0][0]);
        acc[0][1] = WMMA_BF16(a[buf][0], b[buf][1], acc[0][1]);
        acc[1][0] = WMMA_BF16(a[buf][1], b[buf][0], acc[1][0]);
        acc[1][1] = WMMA_BF16(a[buf][1], b[buf][1], acc[1][1]);
    };

    loadF(0, 0);
    for (int kc = 0; kc < KC; kc += 64) {       // KC multiple of 64
        loadF(1, kc + 32);                      // prefetch next chunk
        mmac(0);
        if (kc + 64 < KC) loadF(0, kc + 64);    // prefetch chunk after
        mmac(1);
    }

    const int mh = (lane >> 4) * 8;
#pragma unroll
    for (int mi = 0; mi < 2; ++mi)
#pragma unroll
        for (int ni = 0; ni < 2; ++ni) {
            const int   ncol = n0 + ni * 16 + nl;
            const float bv   = bias[ncol];
#pragma unroll
            for (int r = 0; r < 8; ++r)
                g[(size_t)(m0 + mi * 16 + mh + r) * G4 + ncol] = acc[mi][ni][r] + bv;
        }
}

// =====================================================================
// Kernel FI: pf = c@P_f, pi = c@P_i (dual WMMA GEMM, shared A frags)
// + fused cell update: c_new = c*sig(gf+pf) + tanh(gc)*sig(gi+pi)
// wave tile 32x32 (x2 matrices); block 64x64; grid (4, 16)
// =====================================================================
__global__ __launch_bounds__(128) void lstm_kfi(
    const bf16_t* __restrict__ cBin,  // [B,H] bf16  c_{t-1} (A operand)
    const bf16_t* __restrict__ Pft,   // [H,H] bf16  P_f^T
    const bf16_t* __restrict__ Pit,   // [H,H] bf16  P_i^T
    const float*  __restrict__ g,     // [B,4H]
    float*        __restrict__ cF,    // [B,H] f32   c master (rd old, wr new)
    bf16_t*       __restrict__ cBout) // [B,H] bf16  c_t (other buffer)
{
    const int lane = threadIdx.x & 31;
    const int wave = threadIdx.x >> 5;
    const int m0   = blockIdx.x * 64 + (wave & 1) * 32;
    const int n0   = blockIdx.y * 64 + (wave >> 1) * 32;
    const int nl   = lane & 15;
    const int khA  = (lane >> 4) * 16;
    const int kbB  = (lane >> 4) * 16;

    const bf16_t* ar0 = cBin + (size_t)(m0 + nl) * Hsz + khA;
    const bf16_t* ar1 = cBin + (size_t)(m0 + 16 + nl) * Hsz + khA;
    const size_t  co0 = (size_t)(n0 + nl) * Hsz + kbB;
    const size_t  co1 = (size_t)(n0 + 16 + nl) * Hsz + kbB;

    v8f accF[2][2] = {};
    v8f accI[2][2] = {};
    v16bf a[2][2], bf[2][2], bi[2][2];

    auto loadF = [&](int buf, int kc) {
        a[buf][0]  = ld_fragA(ar0 + kc);
        a[buf][1]  = ld_fragA(ar1 + kc);
        bf[buf][0] = ld_fragB(Pft + co0 + kc);
        bf[buf][1] = ld_fragB(Pft + co1 + kc);
        bi[buf][0] = ld_fragB(Pit + co0 + kc);
        bi[buf][1] = ld_fragB(Pit + co1 + kc);
    };
    auto mmac = [&](int buf) {
        accF[0][0] = WMMA_BF16(a[buf][0], bf[buf][0], accF[0][0]);
        accF[0][1] = WMMA_BF16(a[buf][0], bf[buf][1], accF[0][1]);
        accF[1][0] = WMMA_BF16(a[buf][1], bf[buf][0], accF[1][0]);
        accF[1][1] = WMMA_BF16(a[buf][1], bf[buf][1], accF[1][1]);
        accI[0][0] = WMMA_BF16(a[buf][0], bi[buf][0], accI[0][0]);
        accI[0][1] = WMMA_BF16(a[buf][0], bi[buf][1], accI[0][1]);
        accI[1][0] = WMMA_BF16(a[buf][1], bi[buf][0], accI[1][0]);
        accI[1][1] = WMMA_BF16(a[buf][1], bi[buf][1], accI[1][1]);
    };

    loadF(0, 0);
    for (int kc = 0; kc < Hsz; kc += 64) {      // H multiple of 64
        loadF(1, kc + 32);
        mmac(0);
        if (kc + 64 < Hsz) loadF(0, kc + 64);
        mmac(1);
    }

    const int mh = (lane >> 4) * 8;
#pragma unroll
    for (int mi = 0; mi < 2; ++mi)
#pragma unroll
        for (int ni = 0; ni < 2; ++ni) {
            const int j = n0 + ni * 16 + nl;
#pragma unroll
            for (int r = 0; r < 8; ++r) {
                const int    mm  = m0 + mi * 16 + mh + r;
                const size_t gij = (size_t)mm * G4 + j;
                const float  gfv = g[gij];
                const float  giv = g[gij + Hsz];
                const float  gcv = g[gij + 2 * Hsz];
                const size_t cij = (size_t)mm * Hsz + j;
                const float  co  = cF[cij];
                const float  fg  = sigmoidf_(gfv + accF[mi][ni][r]);
                const float  ig  = sigmoidf_(giv + accI[mi][ni][r]);
                const float  cn  = co * fg + tanhf(gcv) * ig;
                cF[cij]    = cn;
                cBout[cij] = (bf16_t)cn;
            }
        }
}

// =====================================================================
// Kernel O: po = c_new@P_o + fused output gate:
// h = tanh(c_new) * sig(go + po) ; writes h f32 (d_out) + h bf16
// wave tile 32x32; block 64x64; grid (4, 16)
// =====================================================================
__global__ __launch_bounds__(128) void lstm_ko(
    const bf16_t* __restrict__ cBn,   // [B,H] bf16  c_t (A operand)
    const bf16_t* __restrict__ Pot,   // [H,H] bf16  P_o^T
    const float*  __restrict__ g,     // [B,4H]
    const float*  __restrict__ cF,    // [B,H] f32   c_t
    float*        __restrict__ hOut,  // [B,H] f32   (d_out)
    bf16_t*       __restrict__ hB)    // [B,H] bf16  h_t for next step
{
    const int lane = threadIdx.x & 31;
    const int wave = threadIdx.x >> 5;
    const int m0   = blockIdx.x * 64 + (wave & 1) * 32;
    const int n0   = blockIdx.y * 64 + (wave >> 1) * 32;
    const int nl   = lane & 15;
    const int khA  = (lane >> 4) * 16;
    const int kbB  = (lane >> 4) * 16;

    const bf16_t* ar0 = cBn + (size_t)(m0 + nl) * Hsz + khA;
    const bf16_t* ar1 = cBn + (size_t)(m0 + 16 + nl) * Hsz + khA;
    const bf16_t* bc0 = Pot + (size_t)(n0 + nl) * Hsz + kbB;
    const bf16_t* bc1 = Pot + (size_t)(n0 + 16 + nl) * Hsz + kbB;

    v8f acc[2][2] = {};
    v16bf a[2][2], b[2][2];

    auto loadF = [&](int buf, int kc) {
        a[buf][0] = ld_fragA(ar0 + kc);
        a[buf][1] = ld_fragA(ar1 + kc);
        b[buf][0] = ld_fragB(bc0 + kc);
        b[buf][1] = ld_fragB(bc1 + kc);
    };
    auto mmac = [&](int buf) {
        acc[0][0] = WMMA_BF16(a[buf][0], b[buf][0], acc[0][0]);
        acc[0][1] = WMMA_BF16(a[buf][0], b[buf][1], acc[0][1]);
        acc[1][0] = WMMA_BF16(a[buf][1], b[buf][0], acc[1][0]);
        acc[1][1] = WMMA_BF16(a[buf][1], b[buf][1], acc[1][1]);
    };

    loadF(0, 0);
    for (int kc = 0; kc < Hsz; kc += 64) {
        loadF(1, kc + 32);
        mmac(0);
        if (kc + 64 < Hsz) loadF(0, kc + 64);
        mmac(1);
    }

    const int mh = (lane >> 4) * 8;
#pragma unroll
    for (int mi = 0; mi < 2; ++mi)
#pragma unroll
        for (int ni = 0; ni < 2; ++ni) {
            const int j = n0 + ni * 16 + nl;
#pragma unroll
            for (int r = 0; r < 8; ++r) {
                const int    mm  = m0 + mi * 16 + mh + r;
                const size_t cij = (size_t)mm * Hsz + j;
                const float  go  = g[(size_t)mm * G4 + 3 * Hsz + j];
                const float  og  = sigmoidf_(go + acc[mi][ni][r]);
                const float  hv  = tanhf(cF[cij]) * og;
                hOut[cij] = hv;
                hB[cij]   = (bf16_t)hv;
            }
        }
}

// ---------------- prep kernels ----------------
__global__ void cvt_f32_bf16(const float* __restrict__ in, bf16_t* __restrict__ out, int n) {
    int i = blockIdx.x * 256 + threadIdx.x;
    if (i < n) out[i] = (bf16_t)in[i];
}

// Wcomb^T[n, k] : k<F -> W_x[k, n] ; else R_h[k-F, n].  grid (KC/256, 4H)
__global__ void build_wcombt(const float* __restrict__ Wx, const float* __restrict__ Rh,
                             bf16_t* __restrict__ Wt) {
    int k = blockIdx.x * 256 + threadIdx.x;  // 0..KC-1
    int n = blockIdx.y;                      // 0..4H-1
    float v = (k < Fsz) ? Wx[(size_t)k * G4 + n] : Rh[(size_t)(k - Fsz) * G4 + n];
    Wt[(size_t)n * KC + k] = (bf16_t)v;
}

// P^T[n, k] = P[k, n]   (H x H).  grid (H/256, H)
__global__ void build_pt(const float* __restrict__ P, bf16_t* __restrict__ Pt) {
    int k = blockIdx.x * 256 + threadIdx.x;
    int n = blockIdx.y;
    Pt[(size_t)n * Hsz + k] = (bf16_t)P[(size_t)k * Hsz + n];
}

__global__ void fill_zero_u32(uint32_t* p, int n) {
    int i = blockIdx.x * 256 + threadIdx.x;
    if (i < n) p[i] = 0u;
}

// ---------------- host launcher ----------------
extern "C" void kernel_launch(void* const* d_in, const int* in_sizes, int n_in,
                              void* d_out, int out_size, void* d_ws, size_t ws_size,
                              hipStream_t stream) {
    const float* x    = (const float*)d_in[0];  // [B,T,F]
    const float* W_x  = (const float*)d_in[1];  // [F,4H]
    const float* bias = (const float*)d_in[2];  // [4H]
    const float* R_h  = (const float*)d_in[3];  // [H,4H]
    const float* P_f  = (const float*)d_in[4];  // [H,H]
    const float* P_i  = (const float*)d_in[5];
    const float* P_o  = (const float*)d_in[6];
    float* hOut = (float*)d_out;                // [B,H]

    // workspace carve-up (all offsets 256B-aligned)
    char* ws = (char*)d_ws;
    size_t o = 0;
    bf16_t* xb  = (bf16_t*)(ws + o); o += (size_t)Bsz * Tsz * Fsz * 2;  // 32 MB
    bf16_t* Wt  = (bf16_t*)(ws + o); o += (size_t)G4 * KC * 2;          // 12.6 MB
    bf16_t* Pft = (bf16_t*)(ws + o); o += (size_t)Hsz * Hsz * 2;
    bf16_t* Pit = (bf16_t*)(ws + o); o += (size_t)Hsz * Hsz * 2;
    bf16_t* Pot = (bf16_t*)(ws + o); o += (size_t)Hsz * Hsz * 2;
    bf16_t* hB  = (bf16_t*)(ws + o); o += (size_t)Bsz * Hsz * 2;
    bf16_t* cB0 = (bf16_t*)(ws + o); o += (size_t)Bsz * Hsz * 2;
    bf16_t* cB1 = (bf16_t*)(ws + o); o += (size_t)Bsz * Hsz * 2;
    float*  cF  = (float*) (ws + o); o += (size_t)Bsz * Hsz * 4;
    float*  g   = (float*) (ws + o); o += (size_t)Bsz * G4 * 4;
    (void)o; (void)ws_size; (void)in_sizes; (void)n_in; (void)out_size;

    // ---- prep: bf16 conversions / transposes / state init ----
    {
        int nx = Bsz * Tsz * Fsz;
        cvt_f32_bf16<<<dim3((nx + 255) / 256), 256, 0, stream>>>(x, xb, nx);
        build_wcombt<<<dim3(KC / 256, G4), 256, 0, stream>>>(W_x, R_h, Wt);
        build_pt<<<dim3(Hsz / 256, Hsz), 256, 0, stream>>>(P_f, Pft);
        build_pt<<<dim3(Hsz / 256, Hsz), 256, 0, stream>>>(P_i, Pit);
        build_pt<<<dim3(Hsz / 256, Hsz), 256, 0, stream>>>(P_o, Pot);
        int nh16 = Bsz * Hsz / 2;   // u32 words in a [B,H] bf16 buffer
        int nh32 = Bsz * Hsz;       // u32 words in a [B,H] f32 buffer
        fill_zero_u32<<<dim3((nh16 + 255) / 256), 256, 0, stream>>>((uint32_t*)hB,  nh16);
        fill_zero_u32<<<dim3((nh16 + 255) / 256), 256, 0, stream>>>((uint32_t*)cB0, nh16);
        fill_zero_u32<<<dim3((nh32 + 255) / 256), 256, 0, stream>>>((uint32_t*)cF,  nh32);
    }

    // ---- recurrence: 3 stream-ordered WMMA kernels per step ----
    bf16_t* cb[2] = { cB0, cB1 };
    const dim3 gridG(Bsz / 64, G4 / 64);   // (4, 64)
    const dim3 gridH(Bsz / 64, Hsz / 64);  // (4, 16)
    for (int t = 0; t < Tsz; ++t) {
        bf16_t* cin  = cb[t & 1];
        bf16_t* cout = cb[(t + 1) & 1];
        lstm_kg <<<gridG, 128, 0, stream>>>(xb, hB, Wt, bias, g, t);
        lstm_kfi<<<gridH, 128, 0, stream>>>(cin, Pft, Pit, g, cF, cout);
        lstm_ko <<<gridH, 128, 0, stream>>>(cout, Pot, g, cF, hOut, hB);
    }
}
